// fpn_rpn_outputs_1133871366796
// MI455X (gfx1250) — matrix operand order repaired and checked by reading, not compiled, not run on hardware
//
#include <hip/hip_runtime.h>
#include <hip/hip_bf16.h>

typedef __attribute__((ext_vector_type(16))) __bf16 v16bf;
typedef __attribute__((ext_vector_type(8)))  float  v8f;

union BFrag { uint4 q[2]; v16bf v; };

#if defined(__gfx1250__) && __has_builtin(__builtin_amdgcn_tensor_load_to_lds)
#define HAVE_TDM 1
#else
#define HAVE_TDM 0
#endif

static __device__ __forceinline__ unsigned short f2bf(float f) {
    unsigned int u = __float_as_uint(f);
    unsigned int r = (u + 0x7FFFu + ((u >> 16) & 1u)) >> 16;   // RNE
    return (unsigned short)r;
}

static __device__ __forceinline__ unsigned int fkey(float f) {
    unsigned int u = __float_as_uint(f);
    return (u & 0x80000000u) ? ~u : (u | 0x80000000u);  // monotonic: bigger float -> bigger key
}

// ---------------------------------------------------------------------------
// Weight transform: conv_w (O=256,I=256,3,3) fp32 -> wbT[k=(ky*3+kx)*256+cin][cout] bf16
// ---------------------------------------------------------------------------
__global__ __launch_bounds__(256) void wtransform_kernel(const float* __restrict__ conv_w,
                                                         unsigned short* __restrict__ wbT) {
    int i = blockIdx.x * 256 + threadIdx.x;        // 2304*256 total
    int cout = i & 255;
    int k    = i >> 8;
    int cin  = k & 255;
    int tap  = k >> 8;
    int ky = tap / 3, kx = tap % 3;
    wbT[i] = f2bf(conv_w[((cout * 256 + cin) * 3 + ky) * 3 + kx]);
}

// ---------------------------------------------------------------------------
// Tensor Data Mover: load one 32x256 bf16 weight tile (16KB contiguous) into
// LDS with padding: 16B pad every 512B  ->  row stride 528B (bank-friendly).
// D# built per CDNA5 ISA.
// ---------------------------------------------------------------------------
#define A_STRIDE 40    // ushorts; 80B rows, 16B aligned
#define B_STRIDE 264   // ushorts; 528B rows (512B data + 16B TDM pad)

#if HAVE_TDM
typedef unsigned int u32x4v __attribute__((ext_vector_type(4)));
typedef int          i32x4v __attribute__((ext_vector_type(4)));
typedef int          i32x8v __attribute__((ext_vector_type(8)));

static __device__ __forceinline__ void tdm_load_b_tile(const unsigned short* gsrc,
                                                       unsigned ldsAddr) {
    unsigned long long ga = (unsigned long long)(size_t)gsrc;
    u32x4v g0;
    g0[0] = 1u;                                   // count=1, user descriptor
    g0[1] = ldsAddr;                              // lds_addr [63:32]
    g0[2] = (unsigned)ga;                         // global_addr lo
    g0[3] = (unsigned)(ga >> 32) | 0x80000000u;   // global_addr hi | type=2 ("image")
    i32x8v g1;
    // data_size=3 (8B), pad_enable, pad_interval=6 (128 dwords), pad_amount=3 (4 dwords)
    g1[0] = (3 << 16) | (1 << 20) | (6 << 22) | (3 << 25);
    g1[1] = (int)(2048u << 16);                   // tensor_dim0 = 2048 (x8B = 16KB), lo16
    g1[2] = (int)(1u << 16);                      // tensor_dim0 hi=0 | tensor_dim1=1 lo16
    g1[3] = (int)(2048u << 16);                   // tensor_dim1 hi=0 | tile_dim0 = 2048
    g1[4] = 0;                                    // tile_dim1=0 (unused), tile_dim2=0
    g1[5] = 2048;                                 // tensor_dim0_stride lo32
    g1[6] = 0;
    g1[7] = 0;
    i32x4v z4 = {0, 0, 0, 0};
#if __clang_major__ >= 23
    i32x8v z8 = {0, 0, 0, 0, 0, 0, 0, 0};
    __builtin_amdgcn_tensor_load_to_lds(g0, g1, z4, z4, z8, 0);
#else
    __builtin_amdgcn_tensor_load_to_lds(g0, g1, z4, z4, 0);
#endif
}
#endif

// ---------------------------------------------------------------------------
// Implicit-GEMM 3x3 conv, bf16 WMMA, fp32 accumulate.
// Workgroup tile: 32 pixels x FULL 256 cout. 8 waves: wm=wave&1 (pixel half),
// wq=wave>>1 (col quarter). Each wave: 1 A fragment reused by 4 WMMAs.
// K = 9 taps * 256 cin = 2304, stepped by 32. B tiles via TDM, double-buffered.
// ---------------------------------------------------------------------------
__global__ __launch_bounds__(256) void conv3x3_wmma_kernel(
    const float* __restrict__ in, const unsigned short* __restrict__ wbT,
    const float* __restrict__ bias, float* __restrict__ h, int H, int W, int wshift) {

    __shared__ __align__(16) unsigned short Als[32 * A_STRIDE];
    __shared__ __align__(16) unsigned short Bls[2][32 * B_STRIDE];

    const int HW = H * W;
    const int mbase = blockIdx.x * 32;
    const int t = threadIdx.x;
    const int wave = t >> 5, lane = t & 31;
    const int wm = wave & 1, wq = wave >> 1;       // 2 x 4 wave grid over 32px x 256ch
    const int g = lane >> 4, lm = lane & 15;
    const int wmask = W - 1;

    v8f acc[4];
    #pragma unroll
    for (int s = 0; s < 4; s++) acc[s] = (v8f){0.f, 0.f, 0.f, 0.f, 0.f, 0.f, 0.f, 0.f};

#if HAVE_TDM
    if (wave == 0) tdm_load_b_tile(wbT, (unsigned)(size_t)&Bls[0][0]);
#endif

    for (int kc = 0; kc < 2304; kc += 32) {
        const int tap = kc >> 8;
        const int ci0 = kc & 255;
        const int dy = tap / 3 - 1, dx = tap % 3 - 1;
#if HAVE_TDM
        const int cur = (kc >> 5) & 1;
#else
        const int cur = 0;
#endif
        __syncthreads();  // all waves done reading LDS from previous iteration

        // --- stage A: 32 pixels x 32 cin, fp32 -> bf16, zero-padded borders
        #pragma unroll
        for (int j = 0; j < 4; j++) {
            int e = t + 256 * j;
            int ci = e >> 5, p = e & 31;
            int m = mbase + p;
            int y = m >> wshift, x = m & wmask;
            int yy = y + dy, xx = x + dx;
            float v = 0.f;
            if (yy >= 0 && yy < H && xx >= 0 && xx < W)
                v = in[(size_t)(ci0 + ci) * HW + (yy << wshift) + xx];
            Als[p * A_STRIDE + ci] = f2bf(v);
        }

#if HAVE_TDM
        // --- double-buffered B via Tensor Data Mover
        if (wave == 0) {
            if (kc + 32 < 2304) {
                tdm_load_b_tile(wbT + (size_t)(kc + 32) * 256,
                                (unsigned)(size_t)&Bls[cur ^ 1][0]);
                __builtin_amdgcn_s_wait_tensorcnt((short)1);  // current tile landed
            } else {
                __builtin_amdgcn_s_wait_tensorcnt((short)0);
            }
        }
#else
        // --- fallback: manual B staging (32 k x 256 n) as 16B chunks
        #pragma unroll
        for (int j = 0; j < 4; j++) {
            int c = t + 256 * j;               // 1024 chunks of 16B
            int row = c >> 5, col = c & 31;
            *(uint4*)&Bls[0][row * B_STRIDE + col * 8] =
                *(const uint4*)(wbT + (size_t)(kc + row) * 256 + col * 8);
        }
#endif
        __syncthreads();

        // --- A fragment (ISA 16-bit A 16x32 layout), reused across 4 WMMAs
        BFrag a;
        const unsigned short* arow = &Als[(wm * 16 + lm) * A_STRIDE];
        a.q[0] = *(const uint4*)(arow + 8 * g);
        a.q[1] = *(const uint4*)(arow + 16 + 8 * g);

        const unsigned short* bbase = &Bls[cur][wq * 64 + lm];  // lane%16 = N column, elements = K
        #pragma unroll
        for (int s = 0; s < 4; s++) {
            BFrag b;
            #pragma unroll
            for (int i = 0; i < 16; i++) {
                int kk = (i < 8) ? (8 * g + i) : (16 + 8 * g + (i - 8));
                b.v[i] = *(const __bf16*)(bbase + kk * B_STRIDE + s * 16);
            }
            acc[s] = __builtin_amdgcn_wmma_f32_16x16x32_bf16(false, a.v, false, b.v,
                                                             (short)0, acc[s], false, false);
        }
    }

    // epilogue: C layout VGPR r, lane L -> M = r + 8*(L/16), N = L%16
    #pragma unroll
    for (int s = 0; s < 4; s++) {
        int n = wq * 64 + s * 16 + lm;
        float bn = bias[n];
        #pragma unroll
        for (int r = 0; r < 8; r++) {
            int m = mbase + wm * 16 + r + 8 * g;
            float v = acc[s][r] + bn;
            h[(size_t)m * 256 + n] = v > 0.f ? v : 0.f;
        }
    }
}

// ---------------------------------------------------------------------------
// Heads: 1x1 convs (cls 6, bbox 12) + sigmoid scores + delta layout
// ---------------------------------------------------------------------------
__global__ __launch_bounds__(256) void heads_kernel(
    const float* __restrict__ h, const float* __restrict__ cls_w, const float* __restrict__ cls_b,
    const float* __restrict__ bbox_w, const float* __restrict__ bbox_b,
    float* __restrict__ scores, float* __restrict__ deltas, int HW) {

    __shared__ float hs[32 * 256];
    __shared__ float outs[32][20];

    const int p0 = blockIdx.x * 32;
    const int t = threadIdx.x;

    #pragma unroll
    for (int j = 0; j < 32; j++) {
        int e = t + 256 * j;
        hs[e] = h[(size_t)p0 * 256 + e];
    }
    __syncthreads();

    const int p = t >> 3, u = t & 7;   // 8 threads per pixel
    for (int o = u; o < 18; o += 8) {
        const float* wr;
        float bb;
        if (o < 6) { wr = cls_w + o * 256; bb = cls_b[o]; }
        else       { wr = bbox_w + (o - 6) * 256; bb = bbox_b[o - 6]; }
        float s = bb;
        const float* hp = &hs[p * 256];
        for (int c = 0; c < 256; c++) s += hp[c] * wr[c];
        outs[p][o] = s;
    }
    __syncthreads();

    // scores: softmax over (2,A) class-1 == sigmoid(c1 - c0); c0=ch a, c1=ch 3+a
    if (t < 32 * 3) {
        int pp = t / 3, a = t - pp * 3;
        float d = outs[pp][3 + a] - outs[pp][a];
        scores[(size_t)(p0 + pp) * 3 + a] = 1.f / (1.f + expf(-d));
    }
    // deltas: row r=(pixel*3+a), col j from bbox channel a*4+j
    for (int e = t; e < 32 * 12; e += 256) {
        int pp = e / 12, rem = e - pp * 12;
        deltas[((size_t)(p0 + pp) * 3 + rem / 4) * 4 + (rem & 3)] = outs[pp][6 + rem];
    }
}

// ---------------------------------------------------------------------------
// Radix select (exact top-K threshold), 4 passes of 8 bits.
// state: [0]=prefix, [1]=need, [2]=cntA, [3]=cntE, [8..263]=hist
// ---------------------------------------------------------------------------
__global__ void rs_init_kernel(int* state, int K) {
    int t = threadIdx.x;
    ((unsigned int*)(state + 8))[t] = 0u;
    if (t == 0) { state[0] = 0; state[1] = K; state[2] = 0; state[3] = 0; }
}

__global__ __launch_bounds__(256) void rs_hist_kernel(const float* __restrict__ s, int N,
                                                      int* state, int pass) {
    int i = blockIdx.x * 256 + threadIdx.x;
    if (i >= N) return;
    unsigned int key = fkey(s[i]);
    bool m = (pass == 0) || ((key >> (32 - 8 * pass)) == (unsigned int)state[0]);
    if (m) atomicAdd((unsigned int*)(state + 8) + ((key >> (24 - 8 * pass)) & 255u), 1u);
}

__global__ void rs_scan_kernel(int* state) {
    unsigned int* hist = (unsigned int*)(state + 8);
    int need = state[1];
    unsigned int cum = 0;
    unsigned int pfx = (unsigned int)state[0];
    for (int b = 255; b >= 0; b--) {
        unsigned int hb = hist[b];
        if (cum + hb >= (unsigned int)need) {
            state[0] = (int)((pfx << 8) | (unsigned int)b);
            state[1] = need - (int)cum;
            break;
        }
        cum += hb;
    }
    for (int b = 0; b < 256; b++) hist[b] = 0;
}

__global__ __launch_bounds__(256) void rs_compact_kernel(const float* __restrict__ s, int N,
                                                         int* state, unsigned int* candKey,
                                                         int* candIdx) {
    int i = blockIdx.x * 256 + threadIdx.x;
    if (i >= N) return;
    unsigned int key = fkey(s[i]);
    unsigned int T = (unsigned int)state[0];
    if (key > T) {
        int pos = atomicAdd(&state[2], 1);
        candKey[pos] = key; candIdx[pos] = i;
    } else if (key == T) {
        int e = atomicAdd(&state[3], 1);
        if (e < state[1]) {
            int pos = atomicAdd(&state[2], 1);
            candKey[pos] = key; candIdx[pos] = i;
        }
    }
}

// ---------------------------------------------------------------------------
// Single-block: bitonic sort (desc key, asc idx), anchor decode + clip, greedy NMS
// ---------------------------------------------------------------------------
__global__ __launch_bounds__(1024) void level_sort_decode_nms_kernel(
    const unsigned int* __restrict__ candKey, const int* __restrict__ candIdx,
    const float* __restrict__ scores, const float* __restrict__ deltas,
    const float* __restrict__ im_info,
    float* __restrict__ out_boxes, float* __restrict__ out_scores,
    int K, int W, float stride, float asize) {

    __shared__ unsigned int k_[1024];
    __shared__ int id_[1024];
    __shared__ float bx1[1024], by1[1024], bx2[1024], by2[1024], bar[1024], bsc[1024];
    __shared__ int act[1024];
    __shared__ int kept;

    const int t = threadIdx.x;
    k_[t]  = (t < K) ? candKey[t] : 0u;
    id_[t] = (t < K) ? candIdx[t] : 0x7FFFFFFF;
    __syncthreads();

    for (int kk = 2; kk <= 1024; kk <<= 1) {
        for (int j = kk >> 1; j > 0; j >>= 1) {
            int ixj = t ^ j;
            if (ixj > t) {
                bool dirDesc = ((t & kk) == 0);
                unsigned int ka = k_[t], kb = k_[ixj];
                int ia = id_[t], ib = id_[ixj];
                bool bBeforeA = (kb > ka) || (kb == ka && ib < ia);
                bool aBeforeB = (ka > kb) || (ka == kb && ia < ib);
                bool sw = dirDesc ? bBeforeA : aBeforeB;
                if (sw) { k_[t] = kb; k_[ixj] = ka; id_[t] = ib; id_[ixj] = ia; }
            }
            __syncthreads();
        }
    }

    const float imh = im_info[0], imw = im_info[1];
    const float BBOX_CLIP = 4.1351666f;  // log(1000/16)
    if (t < K) {
        int r = id_[t];
        int a = r % 3, p = r / 3;
        int y = p / W, x = p - y * W;
        float ratio = (a == 0) ? 0.5f : ((a == 1) ? 1.f : 2.f);
        float ws0 = roundf(sqrtf(stride * stride / ratio));
        float hs0 = roundf(ws0 * ratio);
        float scl = asize / stride;
        float aw = ws0 * scl, ah = hs0 * scl;
        float ctr = 0.5f * (stride - 1.f);
        float ax1 = x * stride + ctr - 0.5f * (aw - 1.f);
        float ay1 = y * stride + ctr - 0.5f * (ah - 1.f);
        float cw = aw, ch = ah;
        float cx = ax1 + 0.5f * cw, cy = ay1 + 0.5f * ch;
        const float* d = deltas + (size_t)r * 4;
        float dxv = d[0], dyv = d[1];
        float dwv = fminf(d[2], BBOX_CLIP), dhv = fminf(d[3], BBOX_CLIP);
        float pcx = dxv * cw + cx, pcy = dyv * ch + cy;
        float pw = expf(dwv) * cw, ph = expf(dhv) * ch;
        float X1 = pcx - 0.5f * pw, Y1 = pcy - 0.5f * ph;
        float X2 = pcx + 0.5f * pw - 1.f, Y2 = pcy + 0.5f * ph - 1.f;
        X1 = fminf(fmaxf(X1, 0.f), imw - 1.f);
        Y1 = fminf(fmaxf(Y1, 0.f), imh - 1.f);
        X2 = fminf(fmaxf(X2, 0.f), imw - 1.f);
        Y2 = fminf(fmaxf(Y2, 0.f), imh - 1.f);
        bx1[t] = X1; by1[t] = Y1; bx2[t] = X2; by2[t] = Y2;
        bar[t] = (X2 - X1 + 1.f) * (Y2 - Y1 + 1.f);
        bsc[t] = scores[r];
        act[t] = 1;
    } else {
        act[t] = 0;
    }
    if (t == 0) kept = 0;
    __syncthreads();

    // greedy NMS over sorted candidates (== argmax order of the reference)
    for (int j = 0; j < K; j++) {
        if (act[j]) {
            if (t == 0) {
                int kk = kept++;
                out_boxes[kk * 4 + 0] = bx1[j];
                out_boxes[kk * 4 + 1] = by1[j];
                out_boxes[kk * 4 + 2] = bx2[j];
                out_boxes[kk * 4 + 3] = by2[j];
                out_scores[kk] = bsc[j];
            }
            float xx1 = bx1[j], yy1 = by1[j], xx2 = bx2[j], yy2 = by2[j], aj = bar[j];
            if (t < K && act[t]) {
                float ix1 = fmaxf(xx1, bx1[t]);
                float iy1 = fmaxf(yy1, by1[t]);
                float ix2 = fminf(xx2, bx2[t]);
                float iy2 = fminf(yy2, by2[t]);
                float iw = fmaxf(0.f, ix2 - ix1 + 1.f);
                float ih = fmaxf(0.f, iy2 - iy1 + 1.f);
                float inter = iw * ih;
                float iou = inter / (aj + bar[t] - inter);
                if (iou > 0.7f || t == j) act[t] = 0;
            }
        }
        __syncthreads();
    }

    // pad to POST_NMS=1000
    int kc = kept;
    if (t < 1000 && t >= kc) {
        out_boxes[t * 4 + 0] = 0.f; out_boxes[t * 4 + 1] = 0.f;
        out_boxes[t * 4 + 2] = 0.f; out_boxes[t * 4 + 3] = 0.f;
        out_scores[t] = -1e9f;
    }
}

// ---------------------------------------------------------------------------
// Final: sort global top-1000 candidates, gather, compute FPN level, write out
// out layout: [0..3999] boxes, [4000..4999] scores, [5000..5999] lvls
// ---------------------------------------------------------------------------
__global__ __launch_bounds__(1024) void final_kernel(
    const unsigned int* __restrict__ candKey, const int* __restrict__ candIdx,
    const float* __restrict__ lvlBoxes, const float* __restrict__ lvlScores,
    float* __restrict__ out) {

    __shared__ unsigned int k_[1024];
    __shared__ int id_[1024];
    const int t = threadIdx.x;
    k_[t]  = (t < 1000) ? candKey[t] : 0u;
    id_[t] = (t < 1000) ? candIdx[t] : 0x7FFFFFFF;
    __syncthreads();

    for (int kk = 2; kk <= 1024; kk <<= 1) {
        for (int j = kk >> 1; j > 0; j >>= 1) {
            int ixj = t ^ j;
            if (ixj > t) {
                bool dirDesc = ((t & kk) == 0);
                unsigned int ka = k_[t], kb = k_[ixj];
                int ia = id_[t], ib = id_[ixj];
                bool bBeforeA = (kb > ka) || (kb == ka && ib < ia);
                bool aBeforeB = (ka > kb) || (ka == kb && ia < ib);
                bool sw = dirDesc ? bBeforeA : aBeforeB;
                if (sw) { k_[t] = kb; k_[ixj] = ka; id_[t] = ib; id_[ixj] = ia; }
            }
            __syncthreads();
        }
    }

    if (t < 1000) {
        int g = id_[t];  // global index into the [5][1000] concatenation
        float b0 = lvlBoxes[g * 4 + 0], b1 = lvlBoxes[g * 4 + 1];
        float b2 = lvlBoxes[g * 4 + 2], b3 = lvlBoxes[g * 4 + 3];
        out[t * 4 + 0] = b0; out[t * 4 + 1] = b1;
        out[t * 4 + 2] = b2; out[t * 4 + 3] = b3;
        out[4000 + t] = lvlScores[g];
        float area = (b2 - b0 + 1.f) * (b3 - b1 + 1.f);
        float scale = sqrtf(fmaxf(area, 1.f));
        float lv = floorf(4.f + log2f(scale / 224.f + 1e-6f));
        lv = fminf(fmaxf(lv, 2.f), 5.f);
        out[5000 + t] = lv;
    }
}

// ---------------------------------------------------------------------------
// Host orchestration
// ---------------------------------------------------------------------------
extern "C" void kernel_launch(void* const* d_in, const int* in_sizes, int n_in,
                              void* d_out, int out_size, void* d_ws, size_t ws_size,
                              hipStream_t stream) {
    (void)in_sizes; (void)n_in; (void)out_size; (void)ws_size;

    // input order: feat_p6, feat_p5, feat_p4, feat_p3, feat_p2, im_info,
    //              conv_w, conv_b, cls_w, cls_b, bbox_w, bbox_b
    const float* feats[5] = {
        (const float*)d_in[4],  // li=0 -> lvl2 -> feat_p2 (256x256)
        (const float*)d_in[3],  // lvl3 (128)
        (const float*)d_in[2],  // lvl4 (64)
        (const float*)d_in[1],  // lvl5 (32)
        (const float*)d_in[0],  // lvl6 (16)
    };
    const float* im_info = (const float*)d_in[5];
    const float* conv_w  = (const float*)d_in[6];
    const float* conv_b  = (const float*)d_in[7];
    const float* cls_w   = (const float*)d_in[8];
    const float* cls_b   = (const float*)d_in[9];
    const float* bbox_w  = (const float*)d_in[10];
    const float* bbox_b  = (const float*)d_in[11];

    char* ws = (char*)d_ws;
    size_t o = 0;
    auto alloc = [&](size_t bytes) {
        void* p = ws + o;
        o += (bytes + 255) & ~size_t(255);
        return p;
    };
    unsigned short* wbT = (unsigned short*)alloc(2304 * 256 * 2);
    float* h        = (float*)alloc((size_t)65536 * 256 * 4);
    float* scores   = (float*)alloc((size_t)196608 * 4);
    float* deltas   = (float*)alloc((size_t)196608 * 16);
    int* state      = (int*)alloc(4096);
    unsigned int* candKey = (unsigned int*)alloc(1024 * 4);
    int* candIdx    = (int*)alloc(1024 * 4);
    float* lvlBoxes = (float*)alloc(5 * 1000 * 4 * 4);
    float* lvlScores= (float*)alloc(5 * 1000 * 4);   // contiguous == concat order

    wtransform_kernel<<<2304 * 256 / 256, 256, 0, stream>>>(conv_w, wbT);

    const int Hs[5] = {256, 128, 64, 32, 16};
    const int Sh[5] = {8, 7, 6, 5, 4};
    for (int li = 0; li < 5; li++) {
        int H = Hs[li], W = H, HW = H * W;
        conv3x3_wmma_kernel<<<HW / 32, 256, 0, stream>>>(feats[li], wbT, conv_b,
                                                         h, H, W, Sh[li]);
        heads_kernel<<<HW / 32, 256, 0, stream>>>(h, cls_w, cls_b, bbox_w, bbox_b,
                                                  scores, deltas, HW);
        int N = HW * 3;
        int K = N < 1000 ? N : 1000;
        rs_init_kernel<<<1, 256, 0, stream>>>(state, K);
        for (int pass = 0; pass < 4; pass++) {
            rs_hist_kernel<<<(N + 255) / 256, 256, 0, stream>>>(scores, N, state, pass);
            rs_scan_kernel<<<1, 1, 0, stream>>>(state);
        }
        rs_compact_kernel<<<(N + 255) / 256, 256, 0, stream>>>(scores, N, state,
                                                               candKey, candIdx);
        float stride = (float)(4 << li);          // 2^(li+2)
        float asize  = 32.f * (float)(1 << li);   // START_SIZE * 2^(lvl-2)
        level_sort_decode_nms_kernel<<<1, 1024, 0, stream>>>(
            candKey, candIdx, scores, deltas, im_info,
            lvlBoxes + (size_t)li * 4000, lvlScores + (size_t)li * 1000,
            K, W, stride, asize);
    }

    // final top-1000 over the 5000 concatenated scores
    rs_init_kernel<<<1, 256, 0, stream>>>(state, 1000);
    for (int pass = 0; pass < 4; pass++) {
        rs_hist_kernel<<<(5000 + 255) / 256, 256, 0, stream>>>(lvlScores, 5000, state, pass);
        rs_scan_kernel<<<1, 1, 0, stream>>>(state);
    }
    rs_compact_kernel<<<(5000 + 255) / 256, 256, 0, stream>>>(lvlScores, 5000, state,
                                                              candKey, candIdx);
    final_kernel<<<1, 1024, 0, stream>>>(candKey, candIdx, lvlBoxes, lvlScores,
                                         (float*)d_out);
}
